// CausalSelfAttention_11622181503120
// MI455X (gfx1250) — matrix operand order repaired and checked by reading, not compiled, not run on hardware
//
#include <hip/hip_runtime.h>

// ---------------------------------------------------------------------------
// CDNA5 (gfx1250) causal self-attention: QKV GEMM -> flash attention -> out GEMM
// Matmuls via v_wmma_f32_16x16x32_bf16 (fp32 accumulate), wave32.
// Tile staging of pure-bf16 copies uses global_load_async_to_lds_b128 (ASYNCcnt).
// ---------------------------------------------------------------------------

typedef __attribute__((ext_vector_type(16))) __bf16       v16bf;
typedef __attribute__((ext_vector_type(8)))  float        v8f;
typedef __attribute__((ext_vector_type(8)))  unsigned int v8u;

constexpr int kB  = 2;
constexpr int kN  = 2048;
constexpr int kD  = 1024;
constexpr int kH  = 16;
constexpr int kHD = 64;
constexpr int kD3 = 3 * kD;

#define DEVINL __device__ __forceinline__

DEVINL unsigned short f2bf(float f) {
  unsigned u = __builtin_bit_cast(unsigned, f);
  unsigned r = u + 0x7FFFu + ((u >> 16) & 1u);   // round-to-nearest-even
  return (unsigned short)(r >> 16);
}

DEVINL v8f vzero() {
  v8f z;
#pragma unroll
  for (int i = 0; i < 8; ++i) z[i] = 0.0f;
  return z;
}

DEVINL v8f wmma_bf16(v8u a, v8u b, v8f c) {
  return __builtin_amdgcn_wmma_f32_16x16x32_bf16(
      false, __builtin_bit_cast(v16bf, a),
      false, __builtin_bit_cast(v16bf, b),
      (short)0, c, false, false);
}

// Async copy of 16 bytes global->LDS (per lane), tracked by ASYNCcnt.
DEVINL void async_b128(unsigned lds_byte_addr, unsigned goff_bytes,
                       const void* sbase) {
  asm volatile("global_load_async_to_lds_b128 %0, %1, %2"
               :
               : "v"(lds_byte_addr), "v"(goff_bytes),
                 "s"((unsigned long long)(size_t)sbase)
               : "memory");
}

DEVINL void wait_async0() {
  asm volatile("s_wait_asynccnt 0" ::: "memory");
}

DEVINL unsigned lds_addr_of(const void* p) {
  return (unsigned)(size_t)p;   // generic LDS addr[31:0] = DS byte offset
}

// ---------------------------------------------------------------------------
// Kernel 1: qkv = x @ W_qkv^T + b_qkv  (scatter into per-head bf16 Q/K/V;
//           1/sqrt(HD) folded into Q).  C[4096,3072] = X[4096,1024]*W^T.
// Block tile 128x128, 256 threads (8 waves as 4x2), wave tile 32x64.
// ---------------------------------------------------------------------------
__global__ __launch_bounds__(256) void qkv_gemm_kernel(
    const float* __restrict__ X, const float* __restrict__ W,
    const float* __restrict__ bias,
    unsigned short* __restrict__ Qb, unsigned short* __restrict__ Kb,
    unsigned short* __restrict__ Vb) {
  __shared__ alignas(16) unsigned short As[128][32];  // [m][k] bf16
  __shared__ alignas(16) unsigned short Bs[128][32];  // [n][k] bf16

  const int tid   = threadIdx.x;
  const int wave  = tid >> 5;
  const int lane  = tid & 31;
  const int lhalf = lane >> 4;
  const int l16   = lane & 15;
  const int waveM = wave >> 1;   // 0..3
  const int waveN = wave & 1;    // 0..1

  const int bn0 = blockIdx.x * 128;
  const int bm0 = blockIdx.y * 128;

  v8f acc[2][4];
#pragma unroll
  for (int rt = 0; rt < 2; ++rt)
#pragma unroll
    for (int nt = 0; nt < 4; ++nt) acc[rt][nt] = vzero();

  for (int k0 = 0; k0 < kD; k0 += 32) {
#pragma unroll
    for (int i = 0; i < 16; ++i) {
      int idx = tid + 256 * i;            // 0..4095
      int m = idx >> 5, kk = idx & 31;
      As[m][kk] = f2bf(X[(size_t)(bm0 + m) * kD + k0 + kk]);
      Bs[m][kk] = f2bf(W[(size_t)(bn0 + m) * kD + k0 + kk]);
    }
    __syncthreads();

    v8u af[2];
#pragma unroll
    for (int rt = 0; rt < 2; ++rt) {
      const unsigned int* ar =
          (const unsigned int*)(&As[waveM * 32 + rt * 16 + l16][0]);
      const int bk = lhalf * 8;
#pragma unroll
      for (int i = 0; i < 4; ++i) {
        af[rt][i]     = ar[(bk + 2 * i) >> 1];
        af[rt][i + 4] = ar[(16 + bk + 2 * i) >> 1];
      }
    }
#pragma unroll
    for (int nt = 0; nt < 4; ++nt) {
      const unsigned int* br =
          (const unsigned int*)(&Bs[waveN * 64 + nt * 16 + l16][0]);
      v8u bf;
#pragma unroll
      for (int i = 0; i < 8; ++i) bf[i] = br[lhalf * 4 + i + (i >> 2) * 4];
      acc[0][nt] = wmma_bf16(af[0], bf, acc[0][nt]);
      acc[1][nt] = wmma_bf16(af[1], bf, acc[1][nt]);
    }
    __syncthreads();
  }

  // Epilogue: bias, Q-scale, scatter to [B,H,N,HD] bf16
#pragma unroll
  for (int nt = 0; nt < 4; ++nt) {
    const int n  = bn0 + waveN * 64 + nt * 16 + l16;
    const float bval = bias[n];
    const int p  = n >> 10;        // 0=q 1=k 2=v
    const int dd = n & 1023;
    const int hh = dd >> 6, hd = dd & 63;
    unsigned short* base = (p == 0) ? Qb : ((p == 1) ? Kb : Vb);
    const float scale = (p == 0) ? 0.125f : 1.0f;   // HD^-0.5 folded into Q
#pragma unroll
    for (int rt = 0; rt < 2; ++rt) {
#pragma unroll
      for (int r = 0; r < 8; ++r) {
        const int m  = bm0 + waveM * 32 + rt * 16 + r + lhalf * 8;
        const int bb = m >> 11, nn = m & 2047;
        float v = (acc[rt][nt][r] + bval) * scale;
        base[((size_t)((bb * kH + hh) * kN + nn)) * kHD + hd] = f2bf(v);
      }
    }
  }
}

// ---------------------------------------------------------------------------
// Kernel 2: flash attention with causal mask, bf16 WMMA, fp32 accum.
// grid = (N/128, B*H); 256 threads; each wave owns 16 q-rows x 64 out cols.
// K tile staged with async global->LDS; V transposed through VGPRs.
// ---------------------------------------------------------------------------
constexpr int ATT_BM = 128;
constexpr int ATT_BK = 32;

__global__ __launch_bounds__(256) void attn_kernel(
    const unsigned short* __restrict__ Qb, const unsigned short* __restrict__ Kb,
    const unsigned short* __restrict__ Vb, unsigned short* __restrict__ Ctx) {
  __shared__ alignas(16) unsigned short lds_k[ATT_BK][kHD];   // [key][d]
  __shared__ alignas(16) unsigned short lds_vt[kHD][ATT_BK];  // [d][key] (V^T)
  __shared__ alignas(16) unsigned short lds_p[8][16][ATT_BK]; // per-wave P

  const int tid   = threadIdx.x;
  const int wave  = tid >> 5;
  const int lane  = tid & 31;
  const int lhalf = lane >> 4;
  const int l16   = lane & 15;

  const int bh = blockIdx.y;
  const int b  = bh >> 4;
  const int h  = bh & 15;
  const int q0   = blockIdx.x * ATT_BM;
  const int qrow = q0 + wave * 16;

  const unsigned short* Q  = Qb + (size_t)bh * kN * kHD;
  const unsigned short* Kp = Kb + (size_t)bh * kN * kHD;
  const unsigned short* Vp = Vb + (size_t)bh * kN * kHD;

  const unsigned ldsK = lds_addr_of(&lds_k[0][0]) + tid * 16;

  // Preload Q A-fragments for this wave's 16 rows (k = d0..31 and d32..63)
  v8u qf[2];
  {
    const unsigned int* qrp = (const unsigned int*)(Q + (size_t)(qrow + l16) * kHD);
    const int bk = lhalf * 8;
#pragma unroll
    for (int i = 0; i < 4; ++i) {
      qf[0][i]     = qrp[(bk + 2 * i) >> 1];
      qf[0][i + 4] = qrp[(16 + bk + 2 * i) >> 1];
      qf[1][i]     = qrp[(32 + bk + 2 * i) >> 1];
      qf[1][i + 4] = qrp[(48 + bk + 2 * i) >> 1];
    }
  }

  v8f o[4];
#pragma unroll
  for (int t = 0; t < 4; ++t) o[t] = vzero();
  float mrun[8], lrun[8];
#pragma unroll
  for (int r = 0; r < 8; ++r) { mrun[r] = -1e30f; lrun[r] = 0.0f; }

  const int kend = q0 + ATT_BM;   // causal: keys beyond block's last row skipped
  for (int kb = 0; kb < kend; kb += ATT_BK) {
    // --- K tile: async global->LDS copy (4KB, 16B per lane) ---
    async_b128(ldsK, (unsigned)(kb * kHD * 2) + tid * 16, Kp);
    // --- V tile: load + transpose into lds_vt through VGPRs ---
    {
      const unsigned int* vs = (const unsigned int*)(Vp + (size_t)kb * kHD);
#pragma unroll
      for (int i = 0; i < 4; ++i) {
        int idx = tid + 256 * i;          // 0..1023 dwords
        int key = idx >> 5;               // 0..31
        int dp  = idx & 31;               // dword within row (d = 2*dp)
        unsigned int w = vs[idx];
        lds_vt[2 * dp][key]     = (unsigned short)(w & 0xffffu);
        lds_vt[2 * dp + 1][key] = (unsigned short)(w >> 16);
      }
    }
    wait_async0();
    __syncthreads();

    // --- S = Q @ K^T : two 16x16 tiles, 2 wmma each (d=64 contraction) ---
    v8f s[2];
#pragma unroll
    for (int t = 0; t < 2; ++t) {
      const unsigned int* krow =
          (const unsigned int*)(&lds_k[t * 16 + l16][0]);
      v8u b0, b1;
#pragma unroll
      for (int i = 0; i < 8; ++i) {
        b0[i] = krow[lhalf * 4 + i + (i >> 2) * 4];        // d 0..31
        b1[i] = krow[16 + lhalf * 4 + i + (i >> 2) * 4];   // d 32..63
      }
      v8f a = vzero();
      a = wmma_bf16(qf[0], b0, a);
      a = wmma_bf16(qf[1], b1, a);
      s[t] = a;
    }

    // --- causal mask + online softmax (rows spread across vgprs) ---
#pragma unroll
    for (int r = 0; r < 8; ++r) {
      const int rowg = qrow + r + lhalf * 8;
      const int c0 = kb + l16;
      const int c1 = kb + 16 + l16;
      float s0v = (c0 <= rowg) ? s[0][r] : -1e30f;
      float s1v = (c1 <= rowg) ? s[1][r] : -1e30f;
      float v = fmaxf(s0v, s1v);
      v = fmaxf(v, __shfl_xor(v, 1, 32));
      v = fmaxf(v, __shfl_xor(v, 2, 32));
      v = fmaxf(v, __shfl_xor(v, 4, 32));
      v = fmaxf(v, __shfl_xor(v, 8, 32));
      const float mnew  = fmaxf(mrun[r], v);
      const float alpha = __expf(mrun[r] - mnew);
      const float p0 = __expf(s0v - mnew);
      const float p1 = __expf(s1v - mnew);
      float rs = p0 + p1;
      rs += __shfl_xor(rs, 1, 32);
      rs += __shfl_xor(rs, 2, 32);
      rs += __shfl_xor(rs, 4, 32);
      rs += __shfl_xor(rs, 8, 32);
      lrun[r] = lrun[r] * alpha + rs;
      mrun[r] = mnew;
#pragma unroll
      for (int t = 0; t < 4; ++t) o[t][r] *= alpha;
      const int lrow = r + lhalf * 8;
      lds_p[wave][lrow][l16]      = f2bf(p0);
      lds_p[wave][lrow][16 + l16] = f2bf(p1);
    }

    // --- re-gather P as A-fragment (wave-local LDS bounce) ---
    v8u pf;
    {
      const unsigned int* prow = (const unsigned int*)(&lds_p[wave][l16][0]);
      const int bk = lhalf * 8;
#pragma unroll
      for (int i = 0; i < 4; ++i) {
        pf[i]     = prow[(bk + 2 * i) >> 1];
        pf[i + 4] = prow[(16 + bk + 2 * i) >> 1];
      }
    }

    // --- O += P @ V : 4 wmma over the 64-wide head dim ---
#pragma unroll
    for (int td = 0; td < 4; ++td) {
      const unsigned int* vrow =
          (const unsigned int*)(&lds_vt[td * 16 + l16][0]);
      v8u bf;
#pragma unroll
      for (int i = 0; i < 8; ++i) bf[i] = vrow[lhalf * 4 + i + (i >> 2) * 4];
      o[td] = wmma_bf16(pf, bf, o[td]);
    }
    __syncthreads();
  }

  // --- finalize: O / l, store bf16 context in [B,N,H*HD] ---
#pragma unroll
  for (int r = 0; r < 8; ++r) {
    const int rowg = qrow + r + lhalf * 8;
    const float inv = 1.0f / lrun[r];
    unsigned short* dst = Ctx + ((size_t)(b * kN + rowg)) * kD + h * kHD;
#pragma unroll
    for (int td = 0; td < 4; ++td) dst[td * 16 + l16] = f2bf(o[td][r] * inv);
  }
}

// ---------------------------------------------------------------------------
// Kernel 3: out = ctx @ W_out^T + b_out  (bf16 A via async LDS copy, fp32 out).
// ---------------------------------------------------------------------------
__global__ __launch_bounds__(256) void out_gemm_kernel(
    const unsigned short* __restrict__ Ctx, const float* __restrict__ W,
    const float* __restrict__ bias, float* __restrict__ Out) {
  __shared__ alignas(16) unsigned short As[128][32];
  __shared__ alignas(16) unsigned short Bs[128][32];

  const int tid   = threadIdx.x;
  const int wave  = tid >> 5;
  const int lane  = tid & 31;
  const int lhalf = lane >> 4;
  const int l16   = lane & 15;
  const int waveM = wave >> 1;
  const int waveN = wave & 1;

  const int bn0 = blockIdx.x * 128;
  const int bm0 = blockIdx.y * 128;

  // per-thread A-tile async mapping: 32B each (row,half)
  const int arow  = tid >> 1;
  const int ahalf = tid & 1;
  const unsigned ldsA = lds_addr_of(&As[0][0]) + arow * 64 + ahalf * 32;

  v8f acc[2][4];
#pragma unroll
  for (int rt = 0; rt < 2; ++rt)
#pragma unroll
    for (int nt = 0; nt < 4; ++nt) acc[rt][nt] = vzero();

  for (int k0 = 0; k0 < kD; k0 += 32) {
    // A tile (bf16): two async b128 copies per lane -> 8KB
    {
      unsigned goff = (unsigned)((bm0 + arow) * (kD * 2) + k0 * 2 + ahalf * 32);
      async_b128(ldsA, goff, Ctx);
      async_b128(ldsA + 16, goff + 16, Ctx);
    }
    // B tile: fp32 -> bf16 conversion through VGPRs
#pragma unroll
    for (int i = 0; i < 16; ++i) {
      int idx = tid + 256 * i;             // 0..4095
      int m = idx >> 5, kk = idx & 31;
      Bs[m][kk] = f2bf(W[(size_t)(bn0 + m) * kD + k0 + kk]);
    }
    wait_async0();
    __syncthreads();

    v8u af[2];
#pragma unroll
    for (int rt = 0; rt < 2; ++rt) {
      const unsigned int* ar =
          (const unsigned int*)(&As[waveM * 32 + rt * 16 + l16][0]);
      const int bk = lhalf * 8;
#pragma unroll
      for (int i = 0; i < 4; ++i) {
        af[rt][i]     = ar[(bk + 2 * i) >> 1];
        af[rt][i + 4] = ar[(16 + bk + 2 * i) >> 1];
      }
    }
#pragma unroll
    for (int nt = 0; nt < 4; ++nt) {
      const unsigned int* br =
          (const unsigned int*)(&Bs[waveN * 64 + nt * 16 + l16][0]);
      v8u bf;
#pragma unroll
      for (int i = 0; i < 8; ++i) bf[i] = br[lhalf * 4 + i + (i >> 2) * 4];
      acc[0][nt] = wmma_bf16(af[0], bf, acc[0][nt]);
      acc[1][nt] = wmma_bf16(af[1], bf, acc[1][nt]);
    }
    __syncthreads();
  }

#pragma unroll
  for (int nt = 0; nt < 4; ++nt) {
    const int n = bn0 + waveN * 64 + nt * 16 + l16;
    const float bval = bias[n];
#pragma unroll
    for (int rt = 0; rt < 2; ++rt) {
#pragma unroll
      for (int r = 0; r < 8; ++r) {
        const int m = bm0 + waveM * 32 + rt * 16 + r + lhalf * 8;
        Out[(size_t)m * kD + n] = acc[rt][nt][r] + bval;
      }
    }
  }
}

// ---------------------------------------------------------------------------
extern "C" void kernel_launch(void* const* d_in, const int* in_sizes, int n_in,
                              void* d_out, int out_size, void* d_ws,
                              size_t ws_size, hipStream_t stream) {
  const float* X    = (const float*)d_in[0];
  const float* Wqkv = (const float*)d_in[1];
  const float* bqkv = (const float*)d_in[2];
  const float* Wout = (const float*)d_in[3];
  const float* bout = (const float*)d_in[4];
  float* Out = (float*)d_out;

  // Workspace (bf16 as u16): Q,K,V [B,H,N,HD] + ctx [B,N,D]  (33.6 MB total)
  unsigned short* ws  = (unsigned short*)d_ws;
  unsigned short* Qb  = ws;
  unsigned short* Kb  = ws + (size_t)kB * kH * kN * kHD;
  unsigned short* Vb  = ws + (size_t)2 * kB * kH * kN * kHD;
  unsigned short* Ctx = ws + (size_t)3 * kB * kH * kN * kHD;

  dim3 blk(256, 1, 1);
  dim3 g1(kD3 / 128, (kB * kN) / 128, 1);   // 24 x 32
  qkv_gemm_kernel<<<g1, blk, 0, stream>>>(X, Wqkv, bqkv, Qb, Kb, Vb);

  dim3 g2(kN / ATT_BM, kB * kH, 1);         // 16 x 32
  attn_kernel<<<g2, blk, 0, stream>>>(Qb, Kb, Vb, Ctx);

  dim3 g3(kD / 128, (kB * kN) / 128, 1);    // 8 x 32
  out_gemm_kernel<<<g3, blk, 0, stream>>>(Ctx, Wout, bout, Out);
}
